// MultiHeadAttention_26362509263191
// MI455X (gfx1250) — matrix operand order repaired and checked by reading, not compiled, not run on hardware
//
#include <hip/hip_runtime.h>
#include <hip/hip_bf16.h>

// MHA forward for B=2,S=2048,D=1024,H=16,dk=64 on gfx1250 (MI455X).
// All GEMMs via v_wmma_f32_16x16x32_f16 (f16 operands, f32 accumulate).
// Workspace: 48 MiB (xh, 4x Wt, Qh, Kh, Vt, Ah), all f16.
// GEMM k-loops fully unrolled (immediate-offset loads, no loop-carried copies);
// attention uses explicit ping-pong K buffers. __launch_bounds__(128,1) for VGPRs.

#define B_  2
#define S_  2048
#define DM  1024
#define H_  16
#define DK  64
#define SCALE_ 0.125f   // 1/sqrt(64)

typedef __attribute__((ext_vector_type(16))) _Float16 v16h;
typedef __attribute__((ext_vector_type(8)))  _Float16 v8h;
typedef __attribute__((ext_vector_type(4)))  _Float16 v4h;
typedef __attribute__((ext_vector_type(8)))  float    v8f;

__device__ __forceinline__ v8f wmma32(v16h a, v16h b, v8f c) {
  // D = A(16x32 f16) * B(32x16 f16) + C(16x16 f32)
  return __builtin_amdgcn_wmma_f32_16x16x32_f16(false, a, false, b, (short)0, c,
                                                false, false);
}

// A-operand (16x32, 16-bit) load, contiguous k order:
// lanes 0-15: row=lane, K = 0..15; lanes 16-31: row=lane-16, K = 16..31.
// p = &mat[row * ld + k0]  (row already uses lane&15)
__device__ __forceinline__ v16h loadA(const _Float16* p, int hi) {
  v8h a0 = *(const v8h*)(p + 16 * hi);
  v8h a1 = *(const v8h*)(p + 16 * hi + 8);
  v16h a;
#pragma unroll
  for (int j = 0; j < 8; ++j) { a[j] = a0[j]; a[j + 8] = a1[j]; }
  return a;
}

__device__ __forceinline__ v8f zero8() {
  v8f z;
#pragma unroll
  for (int j = 0; j < 8; ++j) z[j] = 0.0f;
  return z;
}

// ---------------- stage 0a: x (f32) -> xh (f16) ----------------
__global__ void k_cvt_x(const float* __restrict__ x, _Float16* __restrict__ xh, int n) {
  int i = (blockIdx.x * blockDim.x + threadIdx.x) * 4;
  if (i < n) {
    float4 v = *(const float4*)(x + i);
    v4h o;
    o[0] = (_Float16)v.x; o[1] = (_Float16)v.y;
    o[2] = (_Float16)v.z; o[3] = (_Float16)v.w;
    *(v4h*)(xh + i) = o;
  }
}

// ---------------- stage 0b: W (k,n) f32 -> Wt (n,k) f16, LDS tile transpose ----
__global__ void k_tr_cvt(const float* __restrict__ w, _Float16* __restrict__ wt) {
  __shared__ float tile[32][33];
  const int tx = threadIdx.x & 31, ty = threadIdx.x >> 5;  // 32x8
  const int c0 = blockIdx.x * 32, r0 = blockIdx.y * 32;
#pragma unroll
  for (int r = 0; r < 32; r += 8)
    tile[ty + r][tx] = w[(size_t)(r0 + ty + r) * DM + c0 + tx];
  __syncthreads();
#pragma unroll
  for (int r = 0; r < 32; r += 8)
    wt[(size_t)(c0 + ty + r) * DM + r0 + tx] = (_Float16)tile[tx][ty + r];
}

// ---------------- stage 1: fused QKV projection ----------------
// z = blockIdx.z selects Q/K/V. Wave tile: 32 rows x 64 cols; k-loop fully
// unrolled -> all loads are base+immediate, no address arithmetic in the body.
// Q,K stored (b,h,s,dk) f16 (Q pre-scaled by 1/sqrt(dk)); V stored transposed (b,h,dk,s).
__global__ void __launch_bounds__(128, 1)
k_gemm_qkv(const _Float16* __restrict__ xh,
           const _Float16* __restrict__ WqT,
           const _Float16* __restrict__ WkT,
           const _Float16* __restrict__ WvT,
           const float* __restrict__ bq,
           const float* __restrict__ bk,
           const float* __restrict__ bv,
           _Float16* __restrict__ Qh,
           _Float16* __restrict__ Kh,
           _Float16* __restrict__ Vt) {
  const int lane = threadIdx.x & 31;
  const int w    = threadIdx.x >> 5;
  const int lo   = lane & 15, hi = lane >> 4;
  const int rb   = blockIdx.x * 32;                // token tile base
  const int cb   = blockIdx.y * 256 + w * 64;      // output-channel tile base
  const int z    = blockIdx.z;                     // 0=Q 1=K 2=V

  const _Float16* WT   = (z == 0) ? WqT : (z == 1) ? WkT : WvT;
  const float*    bias = (z == 0) ? bq  : (z == 1) ? bk  : bv;

  const _Float16* arow0 = xh + (size_t)(rb + lo) * DM;
  const _Float16* arow1 = xh + (size_t)(rb + 16 + lo) * DM;
  const _Float16* wrow0 = WT + (size_t)(cb + lo) * DM + 16 * hi;
  const _Float16* wrow1 = WT + (size_t)(cb + 16 + lo) * DM + 16 * hi;
  const _Float16* wrow2 = WT + (size_t)(cb + 32 + lo) * DM + 16 * hi;
  const _Float16* wrow3 = WT + (size_t)(cb + 48 + lo) * DM + 16 * hi;

  v8f acc[2][4];
#pragma unroll
  for (int mi = 0; mi < 2; ++mi)
#pragma unroll
    for (int n = 0; n < 4; ++n) acc[mi][n] = zero8();

#pragma unroll
  for (int kk = 0; kk < DM; kk += 32) {
    v16h a0 = loadA(arow0 + kk, hi);
    v16h a1 = loadA(arow1 + kk, hi);
    v16h b0 = *(const v16h*)(wrow0 + kk);
    v16h b1 = *(const v16h*)(wrow1 + kk);
    v16h b2 = *(const v16h*)(wrow2 + kk);
    v16h b3 = *(const v16h*)(wrow3 + kk);
    acc[0][0] = wmma32(a0, b0, acc[0][0]);
    acc[1][0] = wmma32(a1, b0, acc[1][0]);
    acc[0][1] = wmma32(a0, b1, acc[0][1]);
    acc[1][1] = wmma32(a1, b1, acc[1][1]);
    acc[0][2] = wmma32(a0, b2, acc[0][2]);
    acc[1][2] = wmma32(a1, b2, acc[1][2]);
    acc[0][3] = wmma32(a0, b3, acc[0][3]);
    acc[1][3] = wmma32(a1, b3, acc[1][3]);
  }

#pragma unroll
  for (int n = 0; n < 4; ++n) {
    const int   c  = cb + 16 * n + lo;
    const float bc = bias[c];
    const int   hh = c >> 6, d = c & 63;
#pragma unroll
    for (int mi = 0; mi < 2; ++mi) {
      const int t0 = rb + 16 * mi + 8 * hi;        // token index of vgpr row 0
      const int bb = t0 >> 11, s0 = t0 & (S_ - 1); // rb multiple of 32 -> same batch
      if (z == 2) {
        // V transposed: row d, consecutive s -> packed 16B store
        v8h pk;
#pragma unroll
        for (int r = 0; r < 8; ++r) pk[r] = (_Float16)(acc[mi][n][r] + bc);
        *(v8h*)(Vt + ((size_t)(bb * H_ + hh) * DK + d) * S_ + s0) = pk;
      } else {
        _Float16* dst = (z == 0) ? Qh : Kh;
        const float sc = (z == 0) ? SCALE_ : 1.0f;
#pragma unroll
        for (int r = 0; r < 8; ++r)
          dst[((size_t)(bb * H_ + hh) * S_ + (s0 + r)) * DK + d] =
              (_Float16)((acc[mi][n][r] + bc) * sc);
      }
    }
  }
}

// ---------------- stage 2: flash-style attention ----------------
__device__ __forceinline__ void loadK4(const _Float16* Kb0, const _Float16* Kb1,
                                       int kb, int hi, v16h (&kA)[4]) {
  kA[0] = loadA(Kb0 + (size_t)kb * DK,      hi);
  kA[1] = loadA(Kb0 + (size_t)kb * DK + 32, hi);
  kA[2] = loadA(Kb1 + (size_t)kb * DK,      hi);
  kA[3] = loadA(Kb1 + (size_t)kb * DK + 32, hi);
}

// One 32-key block: S^T = K*Q^T (two 16x16 tiles), online softmax, acc += P*V.
// S^T D-layout maps lane-for-lane onto the f16 A-operand layout of the P*V WMMA.
__device__ __forceinline__ void attn_block(
    const v16h (&kA)[4], int kb, v16h qB0, v16h qB1,
    const _Float16* vr0, const _Float16* vr1,
    const _Float16* vr2, const _Float16* vr3,
    const int* mk0, const int* mk1,
    int hi, float& mrow, float& lsum, v8f (&acc)[4]) {
  // V operands for this block (consumed only after softmax -> latency hidden)
  v16h vB0 = __builtin_shufflevector(*(const v8h*)(vr0 + kb), *(const v8h*)(vr0 + kb + 16), 0,1,2,3,4,5,6,7,8,9,10,11,12,13,14,15);
  v16h vB1 = __builtin_shufflevector(*(const v8h*)(vr1 + kb), *(const v8h*)(vr1 + kb + 16), 0,1,2,3,4,5,6,7,8,9,10,11,12,13,14,15);
  v16h vB2 = __builtin_shufflevector(*(const v8h*)(vr2 + kb), *(const v8h*)(vr2 + kb + 16), 0,1,2,3,4,5,6,7,8,9,10,11,12,13,14,15);
  v16h vB3 = __builtin_shufflevector(*(const v8h*)(vr3 + kb), *(const v8h*)(vr3 + kb + 16), 0,1,2,3,4,5,6,7,8,9,10,11,12,13,14,15);
  int mv0[8], mv1[8];
#pragma unroll
  for (int r = 0; r < 8; ++r) { mv0[r] = mk0[kb + r]; mv1[r] = mk1[kb + r]; }

  v8f st0 = wmma32(kA[0], qB0, zero8());  st0 = wmma32(kA[1], qB1, st0);
  v8f st1 = wmma32(kA[2], qB0, zero8());  st1 = wmma32(kA[3], qB1, st1);

#pragma unroll
  for (int r = 0; r < 8; ++r) {
    if (mv0[r] == 0) st0[r] = -1e30f;
    if (mv1[r] == 0) st1[r] = -1e30f;
  }

  float lmax = -1e30f;
#pragma unroll
  for (int r = 0; r < 8; ++r) lmax = fmaxf(lmax, fmaxf(st0[r], st1[r]));
  lmax = fmaxf(lmax, __shfl_xor(lmax, 16, 32));
  const float mnew  = fmaxf(mrow, lmax);
  const float alpha = __expf(mrow - mnew);
  float psum = 0.0f;
#pragma unroll
  for (int r = 0; r < 8; ++r) {
    st0[r] = __expf(st0[r] - mnew);
    st1[r] = __expf(st1[r] - mnew);
    psum += st0[r] + st1[r];
  }
  psum += __shfl_xor(psum, 16, 32);
  lsum = lsum * alpha + psum;
  mrow = mnew;

  v16h pA;
#pragma unroll
  for (int r = 0; r < 8; ++r) {
    pA[r]     = (_Float16)st0[r];
    pA[r + 8] = (_Float16)st1[r];
  }

  float aq[8];
#pragma unroll
  for (int r = 0; r < 8; ++r) aq[r] = __shfl(alpha, r + 8 * hi, 32);

  v8f c0 = acc[0], c1 = acc[1], c2 = acc[2], c3 = acc[3];
#pragma unroll
  for (int r = 0; r < 8; ++r) {
    c0[r] *= aq[r]; c1[r] *= aq[r]; c2[r] *= aq[r]; c3[r] *= aq[r];
  }
  acc[0] = wmma32(pA, vB0, c0);
  acc[1] = wmma32(pA, vB1, c1);
  acc[2] = wmma32(pA, vB2, c2);
  acc[3] = wmma32(pA, vB3, c3);
}

// 1 wave = 16 q rows; block = 4 waves (64 q rows); grid (S/64, H, B).
// Ping-pong K buffers: kA0 consumed while kA1 loads, then reloaded while kA1's
// block computes -> one block of compute between every K load and its use.
__global__ void __launch_bounds__(128, 1)
k_attn(const _Float16* __restrict__ Qh,
       const _Float16* __restrict__ Kh,
       const _Float16* __restrict__ Vt,
       const int* __restrict__ mask,
       _Float16* __restrict__ Ah) {
  const int lane = threadIdx.x & 31;
  const int w    = threadIdx.x >> 5;
  const int lo   = lane & 15, hi = lane >> 4;
  const int h = blockIdx.y, b = blockIdx.z;
  const int qb = blockIdx.x * 64 + w * 16;

  const _Float16* Qp = Qh + (size_t)(b * H_ + h) * S_ * DK;
  const _Float16* Kp = Kh + (size_t)(b * H_ + h) * S_ * DK;
  const _Float16* Vp = Vt + (size_t)(b * H_ + h) * DK * S_;

  const _Float16* Kb0 = Kp + (size_t)lo * DK;
  const _Float16* Kb1 = Kp + (size_t)(16 + lo) * DK;
  const _Float16* vr0 = Vp + (size_t)(lo)      * S_ + 8 * hi;
  const _Float16* vr1 = Vp + (size_t)(16 + lo) * S_ + 8 * hi;
  const _Float16* vr2 = Vp + (size_t)(32 + lo) * S_ + 8 * hi;
  const _Float16* vr3 = Vp + (size_t)(48 + lo) * S_ + 8 * hi;
  const int* mk0 = mask + b * S_ + 8 * hi;
  const int* mk1 = mask + b * S_ + 16 + 8 * hi;

  // Q^T B-operands (loop invariant): lane=q row, K = dk 16*hi..16*hi+15
  const v16h qB0 = *(const v16h*)(Qp + (size_t)(qb + lo) * DK + 16 * hi);
  const v16h qB1 = *(const v16h*)(Qp + (size_t)(qb + lo) * DK + 32 + 16 * hi);

  float mrow = -1e30f, lsum = 0.0f;
  v8f acc[4];
#pragma unroll
  for (int n = 0; n < 4; ++n) acc[n] = zero8();

  v16h kA0[4], kA1[4];
  loadK4(Kb0, Kb1, 0, hi, kA0);

  for (int kb = 0; kb < S_; kb += 64) {
    loadK4(Kb0, Kb1, kb + 32, hi, kA1);
    attn_block(kA0, kb, qB0, qB1, vr0, vr1, vr2, vr3, mk0, mk1,
               hi, mrow, lsum, acc);
    const int kn = (kb + 64 < S_) ? (kb + 64) : 0;  // clamped (last reload unused)
    loadK4(Kb0, Kb1, kn, hi, kA0);
    attn_block(kA1, kb + 32, qB0, qB1, vr0, vr1, vr2, vr3, mk0, mk1,
               hi, mrow, lsum, acc);
  }

  float linv[8];
#pragma unroll
  for (int r = 0; r < 8; ++r) linv[r] = 1.0f / __shfl(lsum, r + 8 * hi, 32);
#pragma unroll
  for (int n = 0; n < 4; ++n)
#pragma unroll
    for (int r = 0; r < 8; ++r)
      Ah[((size_t)b * S_ + (qb + 8 * hi + r)) * DM + h * DK + 16 * n + lo] =
          (_Float16)(acc[n][r] * linv[r]);
}

// ---------------- stage 3: output projection -> f32 d_out ----------------
__global__ void __launch_bounds__(128, 1)
k_gemm_out(const _Float16* __restrict__ Ah,
           const _Float16* __restrict__ WoT,
           const float* __restrict__ bo,
           float* __restrict__ out) {
  const int lane = threadIdx.x & 31;
  const int w    = threadIdx.x >> 5;
  const int lo   = lane & 15, hi = lane >> 4;
  const int rb   = blockIdx.x * 32;
  const int cb   = blockIdx.y * 256 + w * 64;

  const _Float16* arow0 = Ah + (size_t)(rb + lo) * DM;
  const _Float16* arow1 = Ah + (size_t)(rb + 16 + lo) * DM;
  const _Float16* wrow0 = WoT + (size_t)(cb + lo) * DM + 16 * hi;
  const _Float16* wrow1 = WoT + (size_t)(cb + 16 + lo) * DM + 16 * hi;
  const _Float16* wrow2 = WoT + (size_t)(cb + 32 + lo) * DM + 16 * hi;
  const _Float16* wrow3 = WoT + (size_t)(cb + 48 + lo) * DM + 16 * hi;

  v8f acc[2][4];
#pragma unroll
  for (int mi = 0; mi < 2; ++mi)
#pragma unroll
    for (int n = 0; n < 4; ++n) acc[mi][n] = zero8();

#pragma unroll
  for (int kk = 0; kk < DM; kk += 32) {
    v16h a0 = loadA(arow0 + kk, hi);
    v16h a1 = loadA(arow1 + kk, hi);
    v16h b0 = *(const v16h*)(wrow0 + kk);
    v16h b1 = *(const v16h*)(wrow1 + kk);
    v16h b2 = *(const v16h*)(wrow2 + kk);
    v16h b3 = *(const v16h*)(wrow3 + kk);
    acc[0][0] = wmma32(a0, b0, acc[0][0]);
    acc[1][0] = wmma32(a1, b0, acc[1][0]);
    acc[0][1] = wmma32(a0, b1, acc[0][1]);
    acc[1][1] = wmma32(a1, b1, acc[1][1]);
    acc[0][2] = wmma32(a0, b2, acc[0][2]);
    acc[1][2] = wmma32(a1, b2, acc[1][2]);
    acc[0][3] = wmma32(a0, b3, acc[0][3]);
    acc[1][3] = wmma32(a1, b3, acc[1][3]);
  }

#pragma unroll
  for (int n = 0; n < 4; ++n) {
    const int   c  = cb + 16 * n + lo;
    const float bc = bo[c];
#pragma unroll
    for (int mi = 0; mi < 2; ++mi) {
      const int t0 = rb + 16 * mi + 8 * hi;
#pragma unroll
      for (int r = 0; r < 8; ++r)
        out[(size_t)(t0 + r) * DM + c] = acc[mi][n][r] + bc;
    }
  }
}

// ---------------- host launcher ----------------
extern "C" void kernel_launch(void* const* d_in, const int* in_sizes, int n_in,
                              void* d_out, int out_size, void* d_ws, size_t ws_size,
                              hipStream_t stream) {
  const float* x    = (const float*)d_in[0];
  const int*   mask = (const int*)d_in[1];
  const float* Wq   = (const float*)d_in[2];
  const float* bq   = (const float*)d_in[3];
  const float* Wk   = (const float*)d_in[4];
  const float* bk   = (const float*)d_in[5];
  const float* Wv   = (const float*)d_in[6];
  const float* bv   = (const float*)d_in[7];
  const float* Wo   = (const float*)d_in[8];
  const float* bo   = (const float*)d_in[9];
  float* out = (float*)d_out;

  char* ws = (char*)d_ws;
  const size_t MB = (size_t)1 << 20;
  _Float16* xh  = (_Float16*)(ws + 0 * MB);   // 8 MiB  (4096 x 1024)
  _Float16* WqT = (_Float16*)(ws + 8 * MB);   // 2 MiB  (n,k)
  _Float16* WkT = (_Float16*)(ws + 10 * MB);  // 2 MiB
  _Float16* WvT = (_Float16*)(ws + 12 * MB);  // 2 MiB
  _Float16* WoT = (_Float16*)(ws + 14 * MB);  // 2 MiB
  _Float16* Qh  = (_Float16*)(ws + 16 * MB);  // 8 MiB  (b,h,s,dk), pre-scaled
  _Float16* Kh  = (_Float16*)(ws + 24 * MB);  // 8 MiB  (b,h,s,dk)
  _Float16* Vt  = (_Float16*)(ws + 32 * MB);  // 8 MiB  (b,h,dk,s)
  _Float16* Ah  = (_Float16*)(ws + 40 * MB);  // 8 MiB  (b,s,d)  -> total 48 MiB

  k_cvt_x<<<4096, 256, 0, stream>>>(x, xh, B_ * S_ * DM);

  dim3 trg(32, 32);
  k_tr_cvt<<<trg, 256, 0, stream>>>(Wq, WqT);
  k_tr_cvt<<<trg, 256, 0, stream>>>(Wk, WkT);
  k_tr_cvt<<<trg, 256, 0, stream>>>(Wv, WvT);
  k_tr_cvt<<<trg, 256, 0, stream>>>(Wo, WoT);

  k_gemm_qkv<<<dim3(128, 4, 3), 128, 0, stream>>>(xh, WqT, WkT, WvT, bq, bk, bv,
                                                  Qh, Kh, Vt);
  k_attn<<<dim3(S_ / 64, H_, B_), 128, 0, stream>>>(Qh, Kh, Vt, mask, Ah);
  k_gemm_out<<<dim3(128, 4), 128, 0, stream>>>(Ah, WoT, bo, out);
}
